// FNO3DEncoder_37503654428866
// MI455X (gfx1250) — matrix operand order repaired and checked by reading, not compiled, not run on hardware
//
#include <hip/hip_runtime.h>

// ---------------------------------------------------------------------------
// FNO3D encoder for gfx1250 (MI455X): all channel/spectral contractions as
// WMMA bf16->f32 GEMMs. rfft/irfft (64-pt, 32 modes, ortho) are folded into
// precomputed 64x64 real DFT matrices so the whole spectral conv is 3 GEMM
// stages per axis. Residual stream kept fp32; GEMM operands stored bf16.
// DFT kernels run a double-buffered async pipeline: 4 pencils per block,
// GLOBAL_LOAD_ASYNC_TO_LDS_B128 stages the next 4KB tile while the current
// one is transposed in LDS and consumed by v_wmma_f32_16x16x32_bf16.
// ---------------------------------------------------------------------------

typedef __attribute__((ext_vector_type(16))) __bf16 v16bf;
typedef __attribute__((ext_vector_type(8)))  __bf16 v8bf;
typedef __attribute__((ext_vector_type(8)))  float  v8f;
typedef int v4i_t __attribute__((vector_size(4 * sizeof(int))));
typedef __attribute__((address_space(1))) v4i_t* gv4i_p;  // global 128-bit
typedef __attribute__((address_space(3))) v4i_t* lv4i_p;  // LDS 128-bit

#define DEV static __device__ __forceinline__

#if defined(__has_builtin)
#if __has_builtin(__builtin_amdgcn_global_load_async_to_lds_b128) && \
    __has_builtin(__builtin_amdgcn_s_wait_asynccnt)
#define HAVE_ASYNC_LDS 1
#endif
#endif
#ifndef HAVE_ASYNC_LDS
#define HAVE_ASYNC_LDS 0
#endif

// 16B global -> LDS, async (ASYNCcnt) when available, else synchronous copy.
DEV void stage16(const __bf16* src, __bf16* lds_dst) {
#if HAVE_ASYNC_LDS
  __builtin_amdgcn_global_load_async_to_lds_b128(
      (gv4i_p)src, (lv4i_p)lds_dst, 0, 0);
#else
  *(v8bf*)lds_dst = *(const v8bf*)src;
#endif
}

DEV void wait_async0() {
#if HAVE_ASYNC_LDS
  __builtin_amdgcn_s_wait_asynccnt(0);
#endif
}
DEV void wait_async1() {
#if HAVE_ASYNC_LDS
  __builtin_amdgcn_s_wait_asynccnt(1);
#endif
}

DEV v8f wmma_bf(v16bf a, v16bf b, v8f c) {
  // D = A(16x32 bf16) x B(32x16 bf16) + C(16x16 f32)
  return __builtin_amdgcn_wmma_f32_16x16x32_bf16(false, a, false, b,
                                                 (short)0, c, false, false);
}

DEV float gelu_exact(float v) {
  return 0.5f * v * (1.0f + erff(v * 0.70710678118654752f));
}

// A-operand (16x32 bf16) from row-major matrix, contiguous K order used here (B loaders match):
//  lane<16 : M=lane,    K = k0+0..15
//  lane>=16: M=lane-16, K = k0+16..31
// K run is 16 contiguous bf16 -> two 16B vector loads per lane.
DEV v16bf load_a_tile(const __bf16* __restrict__ base, int row0, int lda,
                      int k0, int lane) {
  int m = row0 + (lane & 15);
  int half = lane >> 4;
  const __bf16* p = base + (size_t)m * lda + (size_t)(k0 + half * 16);
  v8bf lo = *(const v8bf*)(p);
  v8bf hi = *(const v8bf*)(p + 8);
  v16bf r;
#pragma unroll
  for (int i = 0; i < 8; ++i) { r[i] = lo[i]; r[8 + i] = hi[i]; }
  return r;
}

// B-operand (32x16 bf16) from a K-contiguous (transposed, [n][k], ldk=64)
// LDS tile. Layout: lane<16 -> N=lane, K=k0+0..15; lane>=16 -> K=k0+16..31.
// 16 contiguous bf16 -> two 16B ds_load_b128 per lane.
DEV v16bf load_b_ldsT(const __bf16* tileT, int n0, int k0, int lane) {
  int n = n0 + (lane & 15);
  int half = lane >> 4;
  const __bf16* p = tileT + (size_t)n * 64 + k0 + half * 16;
  v8bf lo = *(const v8bf*)(p);
  v8bf hi = *(const v8bf*)(p + 8);
  v16bf r;
#pragma unroll
  for (int i = 0; i < 8; ++i) { r[i] = lo[i]; r[8 + i] = hi[i]; }
  return r;
}

// ------------------------- kernels -----------------------------------------

// Precompute forward rDFT (modes packed re[0..31], im[32..63]) and inverse
// matrices, ortho norm, bf16.
__global__ void k_init_f(__bf16* __restrict__ Ff, __bf16* __restrict__ Fi) {
  int id = blockIdx.x * blockDim.x + threadIdx.x;   // 4096
  int a = id >> 6, c = id & 63;
  const float w = 6.283185307179586f / 64.0f;
  // Ffwd[m'][t]:  m'<32 -> cos, m'>=32 -> -sin  (scale 1/sqrt(64))
  {
    int mm = a & 31;
    float ang = w * (float)(mm * c);
    float v = (a < 32) ? (0.125f * cosf(ang)) : (-0.125f * sinf(ang));
    Ff[id] = (__bf16)v;
  }
  // Finv[t][m']:  amp = 1/8 for m=0, 2/8 for m>=1 (Nyquist dropped, M=32)
  {
    int mm = c & 31;
    float ang = w * (float)(mm * a);
    float amp = (mm == 0) ? 0.125f : 0.25f;
    float v = (c < 32) ? (amp * cosf(ang)) : (-amp * sinf(ang));
    Fi[id] = (__bf16)v;
  }
}

// Lift: [x, gx, gy, gz] -> 16 (gelu) -> 32, channels-last out.
__global__ void k_lift(const float* __restrict__ x,
                       const float* __restrict__ lw1, const float* __restrict__ lb1,
                       const float* __restrict__ lw2, const float* __restrict__ lb2,
                       float* __restrict__ h, __bf16* __restrict__ hb) {
  int v = blockIdx.x * blockDim.x + threadIdx.x;    // B*64^3 voxels
  int z = v & 63, y = (v >> 6) & 63, xx = (v >> 12) & 63;
  float feat[4];
  feat[0] = x[v];                                   // CIN==1: same flat index
  feat[1] = (float)xx * (1.0f / 63.0f);
  feat[2] = (float)y  * (1.0f / 63.0f);
  feat[3] = (float)z  * (1.0f / 63.0f);
  float mid[16];
#pragma unroll
  for (int j = 0; j < 16; ++j) {
    float a = lb1[j];
#pragma unroll
    for (int k = 0; k < 4; ++k) a += lw1[j * 4 + k] * feat[k];
    mid[j] = gelu_exact(a);
  }
#pragma unroll
  for (int o = 0; o < 32; ++o) {
    float a = lb2[o];
#pragma unroll
    for (int j = 0; j < 16; ++j) a += lw2[o * 16 + j] * mid[j];
    size_t idx = (size_t)v * 32 + o;
    h[idx] = a;
    hb[idx] = (__bf16)a;
  }
}

#define PPB 4   // pencils per block in the DFT pipeline

// Forward DFT along one axis: per pencil p, G[m',c] = sum_t Ff[m',t]*h[t,c].
// Double-buffered async stage -> LDS transpose -> 8 waves x WMMA tiles.
__global__ void k_dft_fwd(const __bf16* __restrict__ hb,
                          const __bf16* __restrict__ Ff,
                          __bf16* __restrict__ G,
                          int su, int sv, int st) {
  __shared__ __attribute__((aligned(16))) __bf16 raw[2][2048];  // [t][c] stage
  __shared__ __attribute__((aligned(16))) __bf16 tileT[2048];   // [c][t]
  int tid = threadIdx.x;
  int row = tid >> 2, chunk = tid & 3;              // 4 x 16B per tile row
  int p0 = blockIdx.x * PPB;

  // per-thread global source for pencil p
#define SRC_FWD(p)                                                         \
  (hb + (size_t)((p) >> 12) * 8388608 + (size_t)(((p) >> 6) & 63) * su +   \
   (size_t)((p) & 63) * sv + (size_t)row * st + chunk * 8)

  stage16(SRC_FWD(p0), &raw[0][tid * 8]);
  for (int q = 0; q < PPB; ++q) {
    int p = p0 + q;
    if (q + 1 < PPB) { stage16(SRC_FWD(p + 1), &raw[(q + 1) & 1][tid * 8]); wait_async1(); }
    else             { wait_async0(); }
    __syncthreads();
    { // transpose raw[q&1] -> tileT  (1x ds_load_b128 + 8x ds_store_b16)
      v8bf v = *(const v8bf*)&raw[q & 1][tid * 8];
#pragma unroll
      for (int i = 0; i < 8; ++i) tileT[(chunk * 8 + i) * 64 + row] = v[i];
    }
    __syncthreads();

    int wave = tid >> 5, lane = tid & 31;
    int m0 = (wave >> 1) * 16, c0 = (wave & 1) * 16;
    int n = lane & 15, half = lane >> 4;
    v8f acc = {};
#pragma unroll
    for (int k0 = 0; k0 < 64; k0 += 32) {
      v16bf a = load_a_tile(Ff, m0, 64, k0, lane);
      v16bf bm = load_b_ldsT(tileT, c0, k0, lane);
      acc = wmma_bf(a, bm, acc);
    }
    size_t gbase = (size_t)p * 64 * 32;
#pragma unroll
    for (int r = 0; r < 8; ++r) {
      int m = m0 + r + 8 * half;
      G[gbase + (size_t)m * 32 + c0 + n] = (__bf16)acc[r];
    }
  }
#undef SRC_FWD
}

// Per-mode complex channel mix: out = G * (Wr + i Wi), 4 real WMMAs/tile.
__global__ void k_modemix(const __bf16* __restrict__ G1,
                          const float* __restrict__ Wt,
                          __bf16* __restrict__ G2, int l) {
  int gw = blockIdx.x * 8 + (threadIdx.x >> 5);     // 65536 waves
  int lane = threadIdx.x & 31;
  int ot = gw & 1, m = (gw >> 1) & 31, pt = gw >> 6;
  int o0 = ot * 16, p0 = pt * 16;
  int n = lane & 15, half = lane >> 4;
  // A tiles: rows over pencils p, K = 32 input channels (contiguous).
  v16bf ar = load_a_tile(G1 + (size_t)m * 32,        p0, 2048, 0, lane);
  v16bf ai = load_a_tile(G1 + (size_t)(32 + m) * 32, p0, 2048, 0, lane);
  __builtin_prefetch(G1 + ((size_t)(p0 + 16) * 64 + m) * 32, 0, 1);
  // Weight tiles: w[((i)*32+o)*32+m]*2 + {0,1}  (i stride 2048 f32)
  v16bf wr, wi, win;
  const float* wb = Wt + (size_t)l * 65536 + (size_t)(o0 + n) * 64 + (size_t)m * 2;
#pragma unroll
  for (int i = 0; i < 16; ++i) {
    const float* q = wb + (size_t)(half * 16 + i) * 2048;
    float re = q[0], im = q[1];
    wr[i] = (__bf16)re; wi[i] = (__bf16)im; win[i] = (__bf16)(-im);
  }
  v8f accr = {}, acci = {};
  accr = wmma_bf(ar, wr,  accr);
  accr = wmma_bf(ai, win, accr);   // - Gi*Wi
  acci = wmma_bf(ar, wi,  acci);
  acci = wmma_bf(ai, wr,  acci);
#pragma unroll
  for (int r = 0; r < 8; ++r) {
    int prow = p0 + r + 8 * half;
    G2[((size_t)prow * 64 + m)      * 32 + o0 + n] = (__bf16)accr[r];
    G2[((size_t)prow * 64 + 32 + m) * 32 + o0 + n] = (__bf16)acci[r];
  }
}

// Inverse DFT along the axis, accumulating fp32 into s (beta=0 on first axis).
// Per-pencil G2 block is 4KB contiguous; same async double-buffered pipeline.
__global__ void k_dft_inv(const __bf16* __restrict__ G2,
                          const __bf16* __restrict__ Fi,
                          float* __restrict__ s,
                          int su, int sv, int st, int beta) {
  __shared__ __attribute__((aligned(16))) __bf16 raw[2][2048];  // [m'][c]
  __shared__ __attribute__((aligned(16))) __bf16 tileT[2048];   // [c][m']
  int tid = threadIdx.x;
  int row = tid >> 2, chunk = tid & 3;
  int p0 = blockIdx.x * PPB;

  stage16(G2 + (size_t)p0 * 2048 + tid * 8, &raw[0][tid * 8]);
  for (int q = 0; q < PPB; ++q) {
    int p = p0 + q;
    if (q + 1 < PPB) {
      stage16(G2 + (size_t)(p + 1) * 2048 + tid * 8, &raw[(q + 1) & 1][tid * 8]);
      wait_async1();
    } else {
      wait_async0();
    }
    __syncthreads();
    {
      v8bf v = *(const v8bf*)&raw[q & 1][tid * 8];
#pragma unroll
      for (int i = 0; i < 8; ++i) tileT[(chunk * 8 + i) * 64 + row] = v[i];
    }
    __syncthreads();

    int b = p >> 12, u = (p >> 6) & 63, vq = p & 63;
    size_t base = (size_t)b * 8388608 + (size_t)u * su + (size_t)vq * sv;
    int wave = tid >> 5, lane = tid & 31;
    int t0 = (wave >> 1) * 16, c0 = (wave & 1) * 16;
    int n = lane & 15, half = lane >> 4;
    v8f acc = {};
#pragma unroll
    for (int k0 = 0; k0 < 64; k0 += 32) {
      v16bf a = load_a_tile(Fi, t0, 64, k0, lane);
      v16bf bm = load_b_ldsT(tileT, c0, k0, lane);
      acc = wmma_bf(a, bm, acc);
    }
#pragma unroll
    for (int r = 0; r < 8; ++r) {
      int t = t0 + r + 8 * half;
      size_t idx = base + (size_t)t * st + (c0 + n);
      float v = acc[r];
      if (beta) v += s[idx];
      s[idx] = v;
    }
  }
}

__global__ void k_cvt_bf16(const float* __restrict__ s, __bf16* __restrict__ sb) {
  size_t id = (size_t)(blockIdx.x * blockDim.x + threadIdx.x) * 4;
#pragma unroll
  for (int j = 0; j < 4; ++j) sb[id + j] = (__bf16)s[id + j];
}

// MLP stage 1: 32 -> 64, +bias, exact gelu, bf16 out.
__global__ void k_mlp1(const __bf16* __restrict__ sb,
                       const float* __restrict__ mw1, const float* __restrict__ mb1,
                       __bf16* __restrict__ t, int l) {
  int gw = blockIdx.x * 8 + (threadIdx.x >> 5);     // 262144 waves
  int lane = threadIdx.x & 31;
  int ot = gw & 3, rt = gw >> 2;
  int o0 = ot * 16, row0 = rt * 16;
  int n = lane & 15, half = lane >> 4;
  v8f acc = {};
  {
    v16bf a = load_a_tile(sb, row0, 32, 0, lane);
    v16bf bm;
    const float* bp = mw1 + (size_t)l * 2048 + (size_t)(o0 + n) * 32 + half * 16;
#pragma unroll
    for (int i = 0; i < 16; ++i) bm[i] = (__bf16)bp[i];
    acc = wmma_bf(a, bm, acc);
  }
  float bias = mb1[l * 64 + o0 + n];
#pragma unroll
  for (int r = 0; r < 8; ++r) {
    int row = row0 + r + 8 * half;
    float v = gelu_exact(acc[r] + bias);
    t[(size_t)row * 64 + o0 + n] = (__bf16)v;
  }
}

// MLP stage 2: 64 -> 32, +bias, residual add into fp32 stream + bf16 mirror.
__global__ void k_mlp2(const __bf16* __restrict__ t,
                       const float* __restrict__ mw2, const float* __restrict__ mb2,
                       float* __restrict__ h, __bf16* __restrict__ hb, int l) {
  int gw = blockIdx.x * 8 + (threadIdx.x >> 5);     // 131072 waves
  int lane = threadIdx.x & 31;
  int ot = gw & 1, rt = gw >> 1;
  int o0 = ot * 16, row0 = rt * 16;
  int n = lane & 15, half = lane >> 4;
  v8f acc = {};
#pragma unroll
  for (int k0 = 0; k0 < 64; k0 += 32) {
    v16bf a = load_a_tile(t, row0, 64, k0, lane);
    v16bf bm;
    const float* bp = mw2 + (size_t)l * 2048 + (size_t)(o0 + n) * 64 + k0 + half * 16;
#pragma unroll
    for (int i = 0; i < 16; ++i) bm[i] = (__bf16)bp[i];
    acc = wmma_bf(a, bm, acc);
  }
  float bias = mb2[l * 32 + o0 + n];
#pragma unroll
  for (int r = 0; r < 8; ++r) {
    int row = row0 + r + 8 * half;
    size_t idx = (size_t)row * 32 + o0 + n;
    float v = acc[r] + bias + h[idx];
    h[idx] = v;
    hb[idx] = (__bf16)v;
  }
}

// channels-last -> [B,C,X,Y,Z] fp32 output.
__global__ void k_out(const float* __restrict__ h, float* __restrict__ out) {
  size_t id = (size_t)blockIdx.x * blockDim.x + threadIdx.x;  // 33.5M
  int c = (int)(id & 31);
  size_t v = id >> 5;
  size_t spatial = v & 262143;
  size_t b = v >> 18;
  out[b * 8388608 + (size_t)c * 262144 + spatial] = h[id];
}

// ------------------------- host --------------------------------------------

extern "C" void kernel_launch(void* const* d_in, const int* in_sizes, int n_in,
                              void* d_out, int out_size, void* d_ws, size_t ws_size,
                              hipStream_t stream) {
  (void)in_sizes; (void)n_in; (void)out_size; (void)ws_size;
  const float* x   = (const float*)d_in[0];
  const float* lw1 = (const float*)d_in[1];
  const float* lb1 = (const float*)d_in[2];
  const float* lw2 = (const float*)d_in[3];
  const float* lb2 = (const float*)d_in[4];
  const float* wax[3] = { (const float*)d_in[7],    // z axis -> wz
                          (const float*)d_in[6],    // y axis -> wy
                          (const float*)d_in[5] };  // x axis -> wx
  const float* mw1 = (const float*)d_in[8];
  const float* mb1 = (const float*)d_in[9];
  const float* mw2 = (const float*)d_in[10];
  const float* mb2 = (const float*)d_in[11];
  float* out = (float*)d_out;

  // workspace layout (regions reused across phase liveness)
  char* w = (char*)d_ws;
  float*  h  = (float*)(w);                          // 128 MB residual stream
  __bf16* hb = (__bf16*)(w + (size_t)134217728);     //  64 MB bf16 mirror
  float*  s  = (float*)(w + (size_t)201326592);      // 128 MB spectral accum
  __bf16* g1 = (__bf16*)(w + (size_t)335544320);     //  64 MB DFT stage
  __bf16* g2 = (__bf16*)(w + (size_t)402653184);     //  64 MB mixed modes
  __bf16* t  = (__bf16*)(w + (size_t)335544320);     // 128 MB MLP hidden (aliases g1+g2)
  __bf16* sb = (__bf16*)(w + (size_t)469762048);     //  64 MB bf16 of s
  __bf16* Ff = (__bf16*)(w + (size_t)536870912);     //   8 KB fwd DFT matrix
  __bf16* Fi = (__bf16*)(w + (size_t)536879104);     //   8 KB inv DFT matrix

  // per-axis strides in the channels-last [B,X,Y,Z,C] layout (elements)
  const int SU[3] = { 131072, 131072, 2048 };   // z: u=x, y: u=x, x: u=y
  const int SV[3] = { 2048,   32,     32   };   // z: v=y, y: v=z, x: v=z
  const int ST[3] = { 32,     2048,   131072 }; // axis stride

  k_init_f<<<64, 64, 0, stream>>>(Ff, Fi);
  k_lift<<<4096, 256, 0, stream>>>(x, lw1, lb1, lw2, lb2, h, hb);

  for (int l = 0; l < 6; ++l) {
    for (int ax = 0; ax < 3; ++ax) {
      k_dft_fwd<<<4096, 256, 0, stream>>>(hb, Ff, g1, SU[ax], SV[ax], ST[ax]);
      k_modemix<<<8192, 256, 0, stream>>>(g1, wax[ax], g2, l);
      k_dft_inv<<<4096, 256, 0, stream>>>(g2, Fi, s, SU[ax], SV[ax], ST[ax],
                                          ax == 0 ? 0 : 1);
    }
    k_cvt_bf16<<<32768, 256, 0, stream>>>(s, sb);
    k_mlp1<<<32768, 256, 0, stream>>>(sb, mw1, mb1, t, l);
    k_mlp2<<<16384, 256, 0, stream>>>(t, mw2, mb2, h, hb, l);
  }
  k_out<<<131072, 256, 0, stream>>>(h, out);
}
